// MultiHeadGraphConvLayer_23313082482748
// MI455X (gfx1250) — matrix operand order repaired and checked by reading, not compiled, not run on hardware
//
#include <hip/hip_runtime.h>
#include <hip/hip_bf16.h>
#include <cstddef>

// ---------------- problem constants ----------------
#define B_  32
#define N_  200
#define D_  256
#define A_  64
#define C_  16
#define H_  8
#define O_  32
#define NJP 224   // j padded to multiple of 32 (K dim of agg GEMM)
#define NIP 208   // i padded to multiple of 16 (M dim)

// ---------------- workspace layout (bytes) ----------------
#define OFF_XT   ((size_t)0)                     // bf16 [B][D][NJP]      3,670,016 B
#define OFF_WAT  ((size_t)3670016)               // bf16 [A][D]              32,768 B
#define OFF_WCT  ((size_t)3702784)               // bf16 [H][O][D]          131,072 B
#define OFF_ROW  ((size_t)3833856)               // f32  [B*N][H]           204,800 B
#define OFF_COL  ((size_t)4038656)               // f32  [B*N][H]           204,800 B
#define OFF_ATT  ((size_t)4243456)               // bf16 [B][H][NIP][NJP] 23,855,104 B
// total ~26.8 MB

typedef __attribute__((ext_vector_type(16))) __bf16 bf16x16;
typedef __attribute__((ext_vector_type(8)))  __bf16 bf16x8;
typedef __attribute__((ext_vector_type(8)))  float  f32x8;

__device__ __forceinline__ f32x8 wmma_bf16(bf16x16 a, bf16x16 b, f32x8 c) {
  // D = A(16x32 bf16) * B(32x16 bf16) + C(16x16 f32)
  return __builtin_amdgcn_wmma_f32_16x16x32_bf16(false, a, false, b, (short)0, c,
                                                 false, false);
}

// A-fragment: per-lane two 16B chunks at base+0 and base+16 elements
// (caller pre-offsets base by +8 for lanes 16..31)
__device__ __forceinline__ bf16x16 load_a_frag(const __bf16* base) {
  bf16x8 lo = *(const bf16x8*)(base);
  bf16x8 hi = *(const bf16x8*)(base + 16);
  return __builtin_shufflevector(lo, hi, 0,1,2,3,4,5,6,7,8,9,10,11,12,13,14,15);
}

// B-fragment: per-lane two 16B chunks at base+0 and base+16 elements
__device__ __forceinline__ bf16x16 load_b_frag(const __bf16* base) {
  bf16x8 lo = *(const bf16x8*)(base);
  bf16x8 hi = *(const bf16x8*)(base + 16);
  return __builtin_shufflevector(lo, hi, 0,1,2,3,4,5,6,7,8,9,10,11,12,13,14,15);
}

// ---------------------------------------------------------------------------
// K0a: x (B,N,D) f32 -> x_T (B,D,NJP) bf16, zero padded j>=200 (LDS transpose)
// grid (56, B), 256 threads; blockIdx.x = jt*8 + dt
// ---------------------------------------------------------------------------
__global__ void k_transpose_x(const float* __restrict__ x, __bf16* __restrict__ xT) {
  __shared__ float tile[32][33];
  int jt = blockIdx.x >> 3, dt = blockIdx.x & 7, b = blockIdx.y;
  int j0 = jt * 32, d0 = dt * 32;
  int tx = threadIdx.x & 31, ty = threadIdx.x >> 5;  // 32 x 8
  for (int r = 0; r < 4; ++r) {
    int jl = ty + r * 8;
    int j  = j0 + jl;
    float v = 0.f;
    if (j < N_) v = x[((size_t)(b * N_ + j)) * D_ + d0 + tx];
    tile[jl][tx] = v;
  }
  __syncthreads();
  for (int r = 0; r < 4; ++r) {
    int dl = ty + r * 8;
    xT[((size_t)(b * D_) + d0 + dl) * NJP + j0 + tx] = (__bf16)tile[tx][dl];
  }
}

// K0b: W_att (D,A) f32 -> Wa_T (A,D) bf16.  grid 64, 256 threads
__global__ void k_transpose_watt(const float* __restrict__ W_att, __bf16* __restrict__ WaT) {
  int a = blockIdx.x, d = threadIdx.x;
  WaT[(size_t)a * D_ + d] = (__bf16)W_att[(size_t)d * A_ + a];
}

// K0c: W_conv (H,D,O) f32 -> Wc_T (H,O,D) bf16.  grid 256 (h*32+o), 256 threads
__global__ void k_transpose_wconv(const float* __restrict__ W_conv, __bf16* __restrict__ WcT) {
  int ho = blockIdx.x, d = threadIdx.x;
  int h = ho >> 5, o = ho & 31;
  WcT[(size_t)ho * D_ + d] = (__bf16)W_conv[((size_t)h * D_ + d) * O_ + o];
}

// ---------------------------------------------------------------------------
// K1: x_att = relu(x @ W_att + b_att)  (WMMA bf16, M=6400,N=64,K=256)
//     fused row = x_att@Wr, col = x_att@Wc  (tiny, via LDS)
// grid 400 blocks (16 rows each), 128 threads = 4 waves; wave w -> a-tile w
// ---------------------------------------------------------------------------
__global__ void k_xatt_rowcol(const float* __restrict__ x,
                              const __bf16* __restrict__ WaT,
                              const float* __restrict__ b_att,
                              const float* __restrict__ W_fin,
                              float* __restrict__ rowv, float* __restrict__ colv) {
  __shared__ float xl[16][65];
  int M0   = blockIdx.x * 16;
  int wave = threadIdx.x >> 5, lane = threadIdx.x & 31;
  int l16  = lane & 15;
  bool hiL = lane >= 16;
  int a0   = wave * 16;

  f32x8 acc = {};
  const float* xrow = x + (size_t)(M0 + l16) * D_;
  for (int kc = 0; kc < 8; ++kc) {
    int k0 = kc * 32;
    const float* p1 = xrow + k0 + (hiL ? 8 : 0);
    const float* p2 = p1 + 16;
    bf16x16 a;
    for (int e = 0; e < 8; ++e) { a[e] = (__bf16)p1[e]; a[e + 8] = (__bf16)p2[e]; }
    const __bf16* bp = WaT + (size_t)(a0 + l16) * D_ + k0 + (hiL ? 8 : 0);
    acc = wmma_bf16(a, load_b_frag(bp), acc);
  }
  int acol = a0 + l16;
  float ba = b_att[acol];
  for (int r = 0; r < 8; ++r) {
    int m   = r + (hiL ? 8 : 0);
    float v = acc[r] + ba;
    xl[m][acol] = v > 0.f ? v : 0.f;
  }
  __syncthreads();
  if (threadIdx.x < 128) {
    int m = threadIdx.x >> 3, hh = threadIdx.x & 7;
    float sr = 0.f, sc = 0.f;
    for (int aa = 0; aa < A_; ++aa) {
      float v = xl[m][aa];
      sr += v * W_fin[aa * H_ + hh];           // Wr rows [0,64)
      sc += v * W_fin[(A_ + aa) * H_ + hh];    // Wc rows [64,128)
    }
    rowv[(size_t)(M0 + m) * H_ + hh] = sr;
    colv[(size_t)(M0 + m) * H_ + hh] = sc;
  }
}

// ---------------------------------------------------------------------------
// K2: scores + softmax(axis=j) * mask -> att bf16 (b,h,i<NIP,j<NJP)
// grid B*NIP blocks, 256 threads. wave w reduces head h=w over j.
// ---------------------------------------------------------------------------
__global__ void k_scores_softmax(const float* __restrict__ adj,
                                 const float* __restrict__ mask,
                                 const float* __restrict__ soft_mask,
                                 const float* __restrict__ W_fin,
                                 const float* __restrict__ b_fin,
                                 const float* __restrict__ rowv,
                                 const float* __restrict__ colv,
                                 __bf16* __restrict__ att) {
  int b = blockIdx.x / NIP, i = blockIdx.x % NIP;
  if (i >= N_) {  // zero-fill padded i rows for all heads
    for (int idx = threadIdx.x; idx < H_ * NJP; idx += 256) {
      int hh = idx / NJP, j = idx % NJP;
      att[((size_t)(b * H_ + hh) * NIP + i) * NJP + j] = (__bf16)0.f;
    }
    return;
  }
  __shared__ float s_lds[H_][NJP];
  __shared__ float colsh[H_];
  if (threadIdx.x < H_)
    colsh[threadIdx.x] = colv[(size_t)(b * N_ + i) * H_ + threadIdx.x];
  __syncthreads();

  int j = threadIdx.x;
  if (j < NJP) {
    if (j < N_) {
      const float* ap = adj + ((size_t)(b * N_ + i) * N_ + j) * C_;
      float c16[C_];
      for (int c = 0; c < C_; ++c) c16[c] = ap[c];
      float sm = soft_mask[(size_t)(b * N_ + i) * N_ + j];
      for (int hh = 0; hh < H_; ++hh) {
        float s = rowv[(size_t)(b * N_ + j) * H_ + hh] + colsh[hh] + b_fin[hh] + sm;
        for (int c = 0; c < C_; ++c) s += c16[c] * W_fin[(2 * A_ + c) * H_ + hh];
        s_lds[hh][j] = s;
      }
    } else {
      for (int hh = 0; hh < H_; ++hh) s_lds[hh][j] = -1e30f;
    }
  }
  __syncthreads();

  int hh = threadIdx.x >> 5, lane = threadIdx.x & 31;
  float m = -1e30f;
  for (int k = 0; k < 7; ++k) m = fmaxf(m, s_lds[hh][lane + 32 * k]);
  for (int off = 16; off > 0; off >>= 1) m = fmaxf(m, __shfl_xor(m, off, 32));
  float e[7], sum = 0.f;
  for (int k = 0; k < 7; ++k) {
    float v = __expf(s_lds[hh][lane + 32 * k] - m);
    e[k] = v; sum += v;
  }
  for (int off = 16; off > 0; off >>= 1) sum += __shfl_xor(sum, off, 32);
  float inv = 1.f / sum;
  __bf16* op = att + ((size_t)(b * H_ + hh) * NIP + i) * NJP;
  for (int k = 0; k < 7; ++k) {
    int jj = lane + 32 * k;
    float v = 0.f;
    if (jj < N_) v = e[k] * inv * mask[(size_t)(b * N_ + i) * N_ + jj];
    op[jj] = (__bf16)v;
  }
}

// ---------------------------------------------------------------------------
// K3: agg = att @ x  (WMMA, M=16 i-tile, N=256 d, K=224 j)  -> LDS bf16
//     out = agg @ W_conv[h] + b_conv + residual, ReLU      (WMMA, 16x32, K=256)
// grid (13 i-tiles, H, B), 128 threads = 4 waves (wave w owns d in [64w,64w+64))
// ---------------------------------------------------------------------------
__global__ void k_agg_conv(const __bf16* __restrict__ att,
                           const __bf16* __restrict__ xT,
                           const __bf16* __restrict__ WcT,
                           const float*  __restrict__ x,
                           const float*  __restrict__ b_conv,
                           float* __restrict__ out) {
  __shared__ __bf16 agg_lds[16][264];
  int it = blockIdx.x, h = blockIdx.y, b = blockIdx.z;
  int i0 = it * 16;
  int wave = threadIdx.x >> 5, lane = threadIdx.x & 31;
  int l16  = lane & 15;
  bool hiL = lane >= 16;

  // ---- GEMM 1: agg(16 x 256) = att(16 x 224) @ x(224 x 256) ----
  f32x8 acc[4] = {{}, {}, {}, {}};
  const __bf16* arow = att + ((size_t)(b * H_ + h) * NIP + i0 + l16) * NJP + (hiL ? 8 : 0);
  const __bf16* xTb  = xT + (size_t)b * D_ * NJP;
  for (int jc = 0; jc < 7; ++jc) {
    int j0 = jc * 32;
    bf16x16 a = load_a_frag(arow + j0);
    for (int f = 0; f < 4; ++f) {
      int d = wave * 64 + f * 16 + l16;
      const __bf16* bp = xTb + (size_t)d * NJP + j0 + (hiL ? 8 : 0);
      acc[f] = wmma_bf16(a, load_b_frag(bp), acc[f]);
    }
  }
  // spill agg tile to LDS as bf16 (re-layout C-frag -> row-major [i][d])
  for (int f = 0; f < 4; ++f) {
    int col = wave * 64 + f * 16 + l16;
    for (int r = 0; r < 8; ++r) {
      int m = r + (hiL ? 8 : 0);
      agg_lds[m][col] = (__bf16)acc[f][r];
    }
  }
  __syncthreads();

  // ---- GEMM 2 (wave 0): out(16 x 32) = agg(16 x 256) @ Wc_T[h]^T ----
  if (wave == 0) {
    f32x8 oacc[2] = {{}, {}};
    const __bf16* wcb = WcT + (size_t)h * O_ * D_;
    for (int kc = 0; kc < 8; ++kc) {
      int k0 = kc * 32;
      const __bf16* ap = &agg_lds[l16][0] + k0 + (hiL ? 8 : 0);
      bf16x16 a = load_a_frag(ap);
      for (int g = 0; g < 2; ++g) {
        const __bf16* bp = wcb + (size_t)(g * 16 + l16) * D_ + k0 + (hiL ? 8 : 0);
        oacc[g] = wmma_bf16(a, load_b_frag(bp), oacc[g]);
      }
    }
    for (int g = 0; g < 2; ++g) {
      int ocol = g * 16 + l16;
      int oc   = h * O_ + ocol;
      float bc = b_conv[h * O_ + ocol];
      for (int r = 0; r < 8; ++r) {
        int m = r + (hiL ? 8 : 0);
        int i = i0 + m;
        if (i < N_) {
          size_t oidx = (size_t)(b * N_ + i) * D_ + oc;
          float v = x[oidx] + oacc[g][r] + bc;     // residual
          out[oidx] = v > 0.f ? v : 0.f;           // final relu
        }
      }
    }
  }
}

// ---------------------------------------------------------------------------
extern "C" void kernel_launch(void* const* d_in, const int* in_sizes, int n_in,
                              void* d_out, int out_size, void* d_ws, size_t ws_size,
                              hipStream_t stream) {
  (void)in_sizes; (void)n_in; (void)out_size; (void)ws_size;
  const float* x         = (const float*)d_in[0];
  const float* adj       = (const float*)d_in[1];
  const float* mask      = (const float*)d_in[2];
  const float* soft_mask = (const float*)d_in[3];
  const float* W_att     = (const float*)d_in[4];
  const float* b_att     = (const float*)d_in[5];
  const float* W_fin     = (const float*)d_in[6];
  const float* b_fin     = (const float*)d_in[7];
  const float* W_conv    = (const float*)d_in[8];
  const float* b_conv    = (const float*)d_in[9];
  float* out = (float*)d_out;

  char* ws = (char*)d_ws;
  __bf16* xT   = (__bf16*)(ws + OFF_XT);
  __bf16* WaT  = (__bf16*)(ws + OFF_WAT);
  __bf16* WcT  = (__bf16*)(ws + OFF_WCT);
  float*  rowv = (float*)(ws + OFF_ROW);
  float*  colv = (float*)(ws + OFF_COL);
  __bf16* att  = (__bf16*)(ws + OFF_ATT);

  k_transpose_x    <<<dim3(56, B_), 256, 0, stream>>>(x, xT);
  k_transpose_watt <<<A_, D_, 0, stream>>>(W_att, WaT);
  k_transpose_wconv<<<H_ * O_, D_, 0, stream>>>(W_conv, WcT);
  k_xatt_rowcol    <<<(B_ * N_) / 16, 128, 0, stream>>>(x, WaT, b_att, W_fin, rowv, colv);
  k_scores_softmax <<<B_ * NIP, 256, 0, stream>>>(adj, mask, soft_mask, W_fin, b_fin,
                                                  rowv, colv, att);
  k_agg_conv       <<<dim3(NIP / 16, H_, B_), 128, 0, stream>>>(att, xT, WcT, x,
                                                                b_conv, out);
}
